// FeatureLoss_1795296330264
// MI455X (gfx1250) — compile-verified, matrix operand order checked by reading, not run on hardware
//
#include <hip/hip_runtime.h>
#include <math.h>

// ---------------- problem constants ----------------
#define NIMG   8
#define CTOT   256
#define GROUPS 32
#define CG     8          // channels per group
#define H      128
#define W      128
#define HW     (H*W)
#define NBOX   16
#define BG     (NIMG*GROUPS)   // 256 groups total
#define NCH    (NIMG*CTOT)     // 2048 channels total
#define IMG_H  800.0f
#define IMG_W  1333.0f
#define GN_EPS 1e-5f
#define TH     4               // output rows per k_final block
#define NBLK_FINAL (BG * (H/TH))   // 8192

// padded g-tile geometry for branchless im2col
#define GW2    (W+2)           // 130 cols (zero border left/right)
#define GROWS  (TH+2)          // 6 rows (halo)
#define GCHS   11              // 8 real channels + 3 zero channels for K-padding
#define CHSTR  (GROWS*GW2)     // 780 floats per channel

typedef __attribute__((ext_vector_type(16))) _Float16 v16h;
typedef __attribute__((ext_vector_type(8)))  float    v8f;

union V16H { v16h v; _Float16 h[16]; };
union V8F  { v8f  v; float f[8]; };

__device__ __forceinline__ float sigm(float x) { return 1.f / (1.f + expf(-x)); }

// ---------------- masks ----------------
__global__ void k_mask(const float* __restrict__ bboxes,
                       float* __restrict__ mask_fg, float* __restrict__ bg_inv)
{
  __shared__ int   bx[NBOX][4];
  __shared__ float bar[NBOX];
  __shared__ float red[256];
  int b = blockIdx.x, t = threadIdx.x;
  if (t < NBOX) {
    const float* bb = bboxes + (b*NBOX + t)*4;
    int wmin = (int)floorf(bb[0] / IMG_W * (float)W);
    int wmax = (int)ceilf (bb[2] / IMG_W * (float)W);
    int hmin = (int)floorf(bb[1] / IMG_H * (float)H);
    int hmax = (int)ceilf (bb[3] / IMG_H * (float)H);
    bx[t][0] = hmin; bx[t][1] = hmax; bx[t][2] = wmin; bx[t][3] = wmax;
    bar[t] = 1.f / ((float)(hmax + 1 - hmin) * (float)(wmax + 1 - wmin));
  }
  __syncthreads();
  float cnt = 0.f;
  for (int p = t; p < HW; p += 256) {
    int h = p >> 7, w = p & (W-1);
    float m = 0.f;
    #pragma unroll
    for (int k = 0; k < NBOX; ++k)
      if (h >= bx[k][0] && h <= bx[k][1] && w >= bx[k][2] && w <= bx[k][3])
        m = fmaxf(m, bar[k]);
    mask_fg[b*HW + p] = m;
    cnt += (m > 0.f) ? 0.f : 1.f;
  }
  red[t] = cnt; __syncthreads();
  for (int s = 128; s > 0; s >>= 1) { if (t < s) red[t] += red[t+s]; __syncthreads(); }
  if (t == 0) bg_inv[b] = (red[0] > 0.f) ? 1.f / red[0] : 0.f;
}

// ---------------- per-channel row/col means + corners ----------------
__global__ void k_stats1(const float* __restrict__ x,
                         float* __restrict__ xh, float* __restrict__ xw,
                         float* __restrict__ corners)
{
  __shared__ float tb[32][129];
  int ch = blockIdx.x;             // 0..2047
  int t  = threadIdx.x;            // 0..127
  const float* g = x + (size_t)ch * HW;
  float cs = 0.f;
  for (int chunk = 0; chunk < 4; ++chunk) {
    __syncthreads();
    #pragma unroll 4
    for (int r = 0; r < 32; ++r) {
      float v = g[(chunk*32 + r)*W + t];
      tb[r][t] = v;
      cs += v;
    }
    __syncthreads();
    if (t < 32) {
      float s = 0.f;
      for (int w = 0; w < W; ++w) s += tb[t][w];
      xh[(size_t)ch*H + chunk*32 + t] = s * (1.f / (float)W);
    }
  }
  xw[(size_t)ch*W + t] = cs * (1.f / (float)H);
  if (t == 0) {
    corners[ch*4+0] = g[0];
    corners[ch*4+1] = g[W-1];
    corners[ch*4+2] = g[(size_t)(H-1)*W];
    corners[ch*4+3] = g[(size_t)(H-1)*W + (W-1)];
  }
}

// ---------------- channel mix + sigmoids + analytic conv-mean softmax ----------------
__global__ void k_mix(const float* __restrict__ xh, const float* __restrict__ xw,
                      const float* __restrict__ corners,
                      const float* __restrict__ w1, const float* __restrict__ b1,
                      const float* __restrict__ w3, const float* __restrict__ b3,
                      float* __restrict__ sh, float* __restrict__ sw,
                      float* __restrict__ x21)
{
  __shared__ float lw1[64], lb1[CG];
  __shared__ float lxh[CG][H], lxw[CG][W];
  int bg = blockIdx.x, t = threadIdx.x;  // 64 threads
  if (t < 64) lw1[t] = w1[t];
  if (t < CG) lb1[t] = b1[t];
  for (int i = t; i < CG*H; i += 64) {
    ((float*)lxh)[i] = xh[(size_t)bg*CG*H + i];
    ((float*)lxw)[i] = xw[(size_t)bg*CG*W + i];
  }
  __syncthreads();
  for (int l = t; l < 2*H; l += 64) {
    #pragma unroll
    for (int o = 0; o < CG; ++o) {
      float s = lb1[o];
      #pragma unroll
      for (int i = 0; i < CG; ++i)
        s += lw1[o*CG + i] * (l < H ? lxh[i][l] : lxw[i][l - H]);
      float sg = sigm(s);
      if (l < H) sh[(size_t)(bg*CG + o)*H + l]       = sg;
      else       sw[(size_t)(bg*CG + o)*W + (l - H)] = sg;
    }
  }
  if (t == 0) {
    // analytic mean of SAME-padded 3x3 conv via inclusion-exclusion
    float tot[CG], rs0[CG], rs1[CG], cs0[CG], cs1[CG];
    for (int ic = 0; ic < CG; ++ic) {
      float s = 0.f;
      for (int h = 0; h < H; ++h) s += lxh[ic][h];
      tot[ic] = s * (float)W;
      rs0[ic] = lxh[ic][0]   * (float)W;  rs1[ic] = lxh[ic][H-1] * (float)W;
      cs0[ic] = lxw[ic][0]   * (float)H;  cs1[ic] = lxw[ic][W-1] * (float)H;
    }
    float x2m[CG];
    for (int o = 0; o < CG; ++o) {
      float s = 0.f;
      for (int ic = 0; ic < CG; ++ic) {
        const float* cr = corners + (bg*CG + ic)*4; // [ (0,0) (0,W-1) (H-1,0) (H-1,W-1) ]
        for (int ky = 0; ky < 3; ++ky)
          for (int kx = 0; kx < 3; ++kx) {
            int dy = ky - 1, dx = kx - 1;
            float R = tot[ic];
            if (dy == -1) R -= rs1[ic]; else if (dy == 1) R -= rs0[ic];
            if (dx == -1) R -= cs1[ic]; else if (dx == 1) R -= cs0[ic];
            if (dy != 0 && dx != 0) {
              int rr = (dy == -1) ? 1 : 0;
              int cc = (dx == -1) ? 1 : 0;
              R += cr[rr*2 + cc];
            }
            s += w3[((o*CG + ic)*3 + ky)*3 + kx] * R;
          }
      }
      x2m[o] = b3[o] + s * (1.f / (float)HW);
    }
    float mx = x2m[0];
    for (int o = 1; o < CG; ++o) mx = fmaxf(mx, x2m[o]);
    float se = 0.f, ex[CG];
    for (int o = 0; o < CG; ++o) { ex[o] = expf(x2m[o] - mx); se += ex[o]; }
    for (int o = 0; o < CG; ++o) x21[bg*CG + o] = ex[o] / se;
  }
}

// ---------------- gated-tensor mean / invstd per channel ----------------
__global__ void k_stats2(const float* __restrict__ x,
                         const float* __restrict__ sh, const float* __restrict__ sw,
                         float* __restrict__ mu, float* __restrict__ istd)
{
  __shared__ float r1[128], r2[128];
  int ch = blockIdx.x, t = threadIdx.x;
  const float* g   = x  + (size_t)ch*HW;
  const float* shp = sh + (size_t)ch*H;
  float swt = sw[(size_t)ch*W + t];
  float a = 0.f, a2 = 0.f;
  for (int h = 0; h < H; ++h) {
    float v = g[h*W + t] * shp[h] * swt;
    a += v; a2 += v*v;
  }
  r1[t] = a; r2[t] = a2; __syncthreads();
  for (int s = 64; s > 0; s >>= 1) {
    if (t < s) { r1[t] += r1[t+s]; r2[t] += r2[t+s]; }
    __syncthreads();
  }
  if (t == 0) {
    float m   = r1[0] * (1.f / (float)HW);
    float var = r2[0] * (1.f / (float)HW) - m*m;
    mu[ch]   = m;
    istd[ch] = rsqrtf(var + GN_EPS);
  }
}

// ---------------- fused conv(WMMA) + GN + softmax-weights + gate + masked loss ----------------
__global__ void __launch_bounds__(256)
k_final(const float* __restrict__ xT, const float* __restrict__ xS,
        const float* __restrict__ w3T, const float* __restrict__ b3T,
        const float* __restrict__ gnwT, const float* __restrict__ gnbT,
        const float* __restrict__ shT, const float* __restrict__ swT,
        const float* __restrict__ muT, const float* __restrict__ istdT,
        const float* __restrict__ x21T,
        const float* __restrict__ w3S, const float* __restrict__ b3S,
        const float* __restrict__ gnwS, const float* __restrict__ gnbS,
        const float* __restrict__ shS, const float* __restrict__ swS,
        const float* __restrict__ muS, const float* __restrict__ istdS,
        const float* __restrict__ x21S,
        const float* __restrict__ mask_fg, const float* __restrict__ bg_inv,
        float* __restrict__ partials)
{
  __shared__ float gbuf[GCHS*CHSTR];    // 34320 B padded g tile (+3 zero channels)
  __shared__ float featT[CG][TH][W];    // 16384 B teacher features staged for diff
  __shared__ float swb[CG][W];          // 4096 B
  __shared__ float shb[CG][TH];         // 128 B
  __shared__ float prm[7][CG];          // b3, gnw, gnb, mu, istd, x21, x11
  __shared__ float lw3p[CG*96];         // 3072 B zero-padded conv weights (K 72..95 = 0)
  __shared__ float red[256];

  int bg   = blockIdx.x >> 5;      // group 0..255
  int tile = blockIdx.x & 31;      // row tile 0..31
  int h0   = tile * TH;
  int b    = bg >> 5;              // image
  int ch0  = bg * CG;              // base global channel
  int tid  = threadIdx.x;
  int lane = tid & 31;
  int wv   = tid >> 5;             // wave 0..7
  int row  = wv & 3;               // output row in tile
  int half = wv >> 2;              // which 64-col half
  int sel  = lane >> 4;            // 0: k 0-7,16-23; 1: k 8-15,24-31 (of each 32-k step)
  int m    = lane & 15;            // WMMA M (oc) row owned by this lane in A

  float facc = 0.f, bacc = 0.f;

  // zero-fill padding channels (ic = 8..10) once; never overwritten later
  for (int i = tid; i < 3*CHSTR; i += 256) gbuf[CG*CHSTR + i] = 0.f;

  for (int e = 0; e < 2; ++e) {
    const float* x    = e ? xS   : xT;
    const float* w3   = e ? w3S  : w3T;
    const float* b3   = e ? b3S  : b3T;
    const float* gnw  = e ? gnwS : gnwT;
    const float* gnb  = e ? gnbS : gnbT;
    const float* sh   = e ? shS  : shT;
    const float* sw   = e ? swS  : swT;
    const float* mu   = e ? muS  : muT;
    const float* istd = e ? istdS: istdT;
    const float* x21  = e ? x21S : x21T;

    __syncthreads();   // previous pass fully consumed before overwriting LDS
    // ---- cooperative tile loads (zero border cols, zero OOB halo rows) ----
    for (int i = tid; i < CG*GROWS*GW2; i += 256) {
      int c  = i / (GROWS*GW2);
      int rr = (i / GW2) % GROWS;
      int xx = i % GW2;            // padded col, memory col = xx-1
      int hg = h0 + rr - 1;
      float v = 0.f;
      if (hg >= 0 && hg < H && xx >= 1 && xx <= W)
        v = x[(size_t)(ch0 + c)*HW + (size_t)hg*W + (xx - 1)];
      gbuf[i] = v;
    }
    for (int i = tid; i < CG*96; i += 256) {
      int mm = i / 96, kk = i % 96;
      lw3p[i] = (kk < 72) ? w3[mm*72 + kk] : 0.f;
    }
    for (int i = tid; i < CG*W; i += 256) ((float*)swb)[i] = sw[(size_t)bg*CG*W + i];
    if (tid < CG*TH) {
      int c = tid / TH, r = tid % TH;
      shb[c][r] = sh[(size_t)(bg*CG + c)*H + h0 + r];
    }
    if (tid < CG) {
      prm[0][tid] = b3[tid];
      prm[1][tid] = gnw[tid];
      prm[2][tid] = gnb[tid];
      prm[3][tid] = mu[bg*CG + tid];
      prm[4][tid] = istd[bg*CG + tid];
      prm[5][tid] = x21[bg*CG + tid];
    }
    __syncthreads();
    if (tid == 0) {
      // x11 = softmax(gnb)  (x1 spatial mean == gnb exactly)
      float mx = prm[2][0];
      for (int c = 1; c < CG; ++c) mx = fmaxf(mx, prm[2][c]);
      float se = 0.f, ex[CG];
      for (int c = 0; c < CG; ++c) { ex[c] = expf(prm[2][c] - mx); se += ex[c]; }
      for (int c = 0; c < CG; ++c) prm[6][c] = ex[c] / se;
    }
    __syncthreads();

    // ---- A fragments: w3 as (16 x 96) f16; rows 8..15 zero, K 72..95 zero ----
    V16H A[3];
    {
      float mmask = (m < CG) ? 1.f : 0.f;
      int mbase = (m & (CG-1)) * 96;
      #pragma unroll
      for (int q = 0; q < 3; ++q)
        #pragma unroll
        for (int j = 0; j < 16; ++j) {
          int k = 32*q + j + 8*((j >> 3) + sel);
          A[q].h[j] = (_Float16)(lw3p[mbase + k] * mmask);
        }
    }

    int rowbase = row * GW2;       // im2col row offset, tile-invariant
    for (int nt0 = 0; nt0 < 4; ++nt0) {
      int nt = half*4 + nt0;
      int w0 = nt * 16;
      int n  = lane & 15;
      int w  = w0 + n;             // output pixel column (memory coords)

      V8F acc;
      #pragma unroll
      for (int r = 0; r < 8; ++r) acc.f[r] = 0.f;

      #pragma unroll
      for (int q = 0; q < 3; ++q) {
        V16H B;
        #pragma unroll
        for (int j = 0; j < 16; ++j) {
          // k for this (q,j) in both lane halves; constants fold at compile time
          int k0 = 32*q + j + ((j >> 3) << 3);
          int k1 = k0 + 8;
          int o0 = (k0/9)*CHSTR + ((k0%9)/3)*GW2 + (k0%9)%3;
          int o1 = (k1/9)*CHSTR + ((k1%9)/3)*GW2 + (k1%9)%3;
          int o  = sel ? o1 : o0;  // branchless; K>=72 lands in zero channels
          B.h[j] = (_Float16)gbuf[o + rowbase + w];
        }
        acc.v = __builtin_amdgcn_wmma_f32_16x16x32_f16(
                    false, A[q].v, false, B.v, (short)0, acc.v, false, false);
      }

      // ---- epilogue: lanes 0..15 each own one pixel (all 8 oc in acc rows) ----
      if (lane < 16) {
        float gv[CG];
        float wsum = 0.f;
        #pragma unroll
        for (int r = 0; r < CG; ++r) {
          float x2 = acc.f[r] + prm[0][r];
          gv[r] = gbuf[r*CHSTR + (row+1)*GW2 + (w+1)];
          float gated = gv[r] * shb[r][row] * swb[r][w];
          float x1 = (gated - prm[3][r]) * prm[4][r] * prm[1][r] + prm[2][r];
          wsum += prm[6][r] * x2 + prm[5][r] * x1;
        }
        float sg = sigm(wsum);
        if (e == 0) {
          #pragma unroll
          for (int r = 0; r < CG; ++r) featT[r][row][w] = gv[r] * sg;
        } else {
          float fgw = mask_fg[(size_t)b*HW + (size_t)(h0 + row)*W + w];
          float bgw = (fgw > 0.f) ? 0.f : bg_inv[b];
          float d2 = 0.f;
          #pragma unroll
          for (int r = 0; r < CG; ++r) {
            float d = gv[r] * sg - featT[r][row][w];
            d2 += d * d;
          }
          facc += fgw * d2;
          bacc += bgw * d2;
        }
      }
    } // nt
  } // e

  // ---- block reduction of the two accumulators ----
  red[tid] = facc; __syncthreads();
  for (int s = 128; s > 0; s >>= 1) { if (tid < s) red[tid] += red[tid+s]; __syncthreads(); }
  float fsum = red[0]; __syncthreads();
  red[tid] = bacc; __syncthreads();
  for (int s = 128; s > 0; s >>= 1) { if (tid < s) red[tid] += red[tid+s]; __syncthreads(); }
  if (tid == 0) {
    partials[2*blockIdx.x + 0] = fsum;
    partials[2*blockIdx.x + 1] = red[0];
  }
}

// ---------------- final deterministic reduction ----------------
__global__ void k_reduce(const float* __restrict__ partials, int nblk, float* __restrict__ out)
{
  __shared__ float rf[256], rb[256];
  int t = threadIdx.x;
  float f = 0.f, bv = 0.f;
  for (int i = t; i < nblk; i += 256) { f += partials[2*i]; bv += partials[2*i+1]; }
  rf[t] = f; rb[t] = bv; __syncthreads();
  for (int s = 128; s > 0; s >>= 1) {
    if (t < s) { rf[t] += rf[t+s]; rb[t] += rb[t+s]; }
    __syncthreads();
  }
  if (t == 0) out[0] = 0.5f * (rf[0] / (float)NIMG) + 0.1f * (rb[0] / (float)NIMG);
}

// ---------------- host launch ----------------
extern "C" void kernel_launch(void* const* d_in, const int* in_sizes, int n_in,
                              void* d_out, int out_size, void* d_ws, size_t ws_size,
                              hipStream_t stream) {
  (void)in_sizes; (void)n_in; (void)out_size; (void)ws_size;
  const float* preds_S  = (const float*)d_in[0];
  const float* preds_T  = (const float*)d_in[1];
  const float* bboxes   = (const float*)d_in[2];
  const float* tw1 = (const float*)d_in[3];  const float* tb1 = (const float*)d_in[4];
  const float* tw3 = (const float*)d_in[5];  const float* tb3 = (const float*)d_in[6];
  const float* tgw = (const float*)d_in[7];  const float* tgb = (const float*)d_in[8];
  const float* sw1 = (const float*)d_in[9];  const float* sb1 = (const float*)d_in[10];
  const float* sw3 = (const float*)d_in[11]; const float* sb3 = (const float*)d_in[12];
  const float* sgw = (const float*)d_in[13]; const float* sgb = (const float*)d_in[14];

  float* wsf = (float*)d_ws;
  size_t off = 0;
  auto alloc = [&](size_t n) { float* p = wsf + off; off += n; return p; };

  // per-stream scratch (stream 0 = teacher path, 1 = student path)
  float *xh[2], *xw[2], *cor[2], *sh[2], *sw[2], *mu[2], *istd[2], *x21[2];
  for (int e = 0; e < 2; ++e) {
    xh[e]   = alloc((size_t)NCH * H);
    xw[e]   = alloc((size_t)NCH * W);
    cor[e]  = alloc((size_t)NCH * 4);
    sh[e]   = alloc((size_t)NCH * H);
    sw[e]   = alloc((size_t)NCH * W);
    mu[e]   = alloc(NCH);
    istd[e] = alloc(NCH);
    x21[e]  = alloc(BG * CG);
  }
  float* mask_fg  = alloc((size_t)NIMG * HW);
  float* bg_inv   = alloc(NIMG);
  float* partials = alloc((size_t)NBLK_FINAL * 2);

  const float* xin[2] = { preds_T, preds_S };
  const float* W1[2] = { tw1, sw1 }; const float* B1[2] = { tb1, sb1 };
  const float* W3[2] = { tw3, sw3 }; const float* B3[2] = { tb3, sb3 };
  const float* GW[2] = { tgw, sgw }; const float* GB[2] = { tgb, sgb };

  k_mask<<<NIMG, 256, 0, stream>>>(bboxes, mask_fg, bg_inv);
  for (int e = 0; e < 2; ++e)
    k_stats1<<<NCH, 128, 0, stream>>>(xin[e], xh[e], xw[e], cor[e]);
  for (int e = 0; e < 2; ++e)
    k_mix<<<BG, 64, 0, stream>>>(xh[e], xw[e], cor[e], W1[e], B1[e], W3[e], B3[e],
                                 sh[e], sw[e], x21[e]);
  for (int e = 0; e < 2; ++e)
    k_stats2<<<NCH, 128, 0, stream>>>(xin[e], sh[e], sw[e], mu[e], istd[e]);

  k_final<<<NBLK_FINAL, 256, 0, stream>>>(
      preds_T, preds_S,
      W3[0], B3[0], GW[0], GB[0], sh[0], sw[0], mu[0], istd[0], x21[0],
      W3[1], B3[1], GW[1], GB[1], sh[1], sw[1], mu[1], istd[1], x21[1],
      mask_fg, bg_inv, partials);

  k_reduce<<<1, 256, 0, stream>>>(partials, NBLK_FINAL, (float*)d_out);
}
